// PartSamplingAttention_89764816487116
// MI455X (gfx1250) — matrix operand order repaired and checked by reading, not compiled, not run on hardware
//
#include <hip/hip_runtime.h>

// ---------------------------------------------------------------------------
// PartSamplingAttention for MI455X (gfx1250, wave32, WMMA bf16/f32-acc)
// ---------------------------------------------------------------------------
#define B_   16
#define N_   784
#define DIM_ 768
#define H_   12
#define HC_  64
#define PF_  512
#define P_   410
#define PP_  416          // P padded to multiple of 32
#define BN_  (B_ * N_)    // 12544
#define BP_  (B_ * P_)    // 6560
#define SCALE_ 0.125f     // 64^-0.5
#define LN_EPS_ 1e-5f

typedef __attribute__((ext_vector_type(16))) __bf16 v16bf;
typedef __attribute__((ext_vector_type(8)))  float  v8f;

union BF8  { uint4 u; __bf16 h[8]; };
union BF4  { uint2 u; __bf16 h[4]; };
union F4   { float4 f; float a[4]; };

// ---------------------------------------------------------------------------
// helpers
// ---------------------------------------------------------------------------
__device__ inline v8f zero8() {
  v8f z;
#pragma unroll
  for (int i = 0; i < 8; ++i) z[i] = 0.f;
  return z;
}

__device__ inline v8f wmma_bf16(v16bf a, v16bf b, v8f c) {
  return __builtin_amdgcn_wmma_f32_16x16x32_bf16(false, a, false, b,
                                                 (short)0, c, false, false);
}

// A fragment 16x32 bf16: caller supplies this lane's row pointer (row = lane&15).
// lanes 0-15: K base 0; lanes 16-31: K base 16; VGPR j -> K=kb+2j (contiguous)
__device__ inline v16bf frag_a(const __bf16* rowptr) {
  int lane = threadIdx.x & 31;
  int kb = (lane & 16) ? 16 : 0;
  v16bf f;
#pragma unroll
  for (int j = 0; j < 8; ++j) {
    int k = kb + 2 * j;
    f[2 * j]     = rowptr[k];
    f[2 * j + 1] = rowptr[k + 1];
  }
  return f;
}

// B fragment 32x16 bf16 from a K-transposed tile: lane's 16 contiguous bf16
// (lanes 0-15 hold K 0-15, lanes 16-31 hold K 16-31; column = lane&15).
__device__ inline v16bf frag_b(const __bf16* p) {
  v16bf f;
#pragma unroll
  for (int j = 0; j < 16; ++j) f[j] = p[j];
  return f;
}

__device__ inline float wred_max16(float v) {
#pragma unroll
  for (int d = 1; d < 16; d <<= 1) v = fmaxf(v, __shfl_xor(v, d, 32));
  return v;
}
__device__ inline float wred_sum16(float v) {
#pragma unroll
  for (int d = 1; d < 16; d <<= 1) v += __shfl_xor(v, d, 32);
  return v;
}

__device__ inline void atomicMaxF(float* a, float v) {
  if (v >= 0.f) atomicMax((int*)a, __float_as_int(v));
  else          atomicMin((unsigned int*)a, __float_as_uint(v));
}

// ---------------------------------------------------------------------------
// elementwise prep (vectorized x4)
// ---------------------------------------------------------------------------
__global__ void cvt_bf16x4_kernel(const float* __restrict__ s, __bf16* __restrict__ d, int n4) {
  int t = blockIdx.x * blockDim.x + threadIdx.x;
  if (t >= n4) return;
  F4 v; v.f = ((const float4*)s)[t];
  BF4 o;
#pragma unroll
  for (int i = 0; i < 4; ++i) o.h[i] = (__bf16)v.a[i];
  ((uint2*)d)[t] = o.u;
}

__global__ void gather_parts_kernel(const float* __restrict__ parts,
                                    const int* __restrict__ ridx,
                                    __bf16* __restrict__ d) {
  int t = blockIdx.x * blockDim.x + threadIdx.x;        // one per 4 elems
  if (t >= BP_ * (DIM_ / 4)) return;
  int m = t / (DIM_ / 4), c4 = t - m * (DIM_ / 4);
  int b = m / P_, p = m - b * P_;
  F4 v; v.f = ((const float4*)(parts + ((size_t)b * PF_ + ridx[p]) * DIM_))[c4];
  BF4 o;
#pragma unroll
  for (int i = 0; i < 4; ++i) o.h[i] = (__bf16)v.a[i];
  ((uint2*)d)[t] = o.u;
}

__global__ void init_kernel(float* colmax, float* colsum, float* fw_acc,
                            __bf16* kbf, __bf16* vbf) {
  int t = blockIdx.x * blockDim.x + threadIdx.x;
  if (t < B_ * H_ * PP_) { colmax[t] = -INFINITY; colsum[t] = 0.f; }
  if (t < BN_) fw_acc[t] = 0.f;
  // zero K/V pad rows (p in [410,416)) so no garbage/NaN feeds WMMA
  if (t < B_ * H_ * (PP_ - P_) * HC_) {
    int bh = t / ((PP_ - P_) * HC_);
    int r  = (t / HC_) % (PP_ - P_);
    int c  = t % HC_;
    size_t idx = (((size_t)bh * PP_) + P_ + r) * HC_ + c;
    kbf[idx] = (__bf16)0.f;
    vbf[idx] = (__bf16)0.f;
  }
}

// ---------------------------------------------------------------------------
// bf16 WMMA GEMM: C[M,Nc] = A[M,K] @ W[K,Nc] + bias, with layout epilogues
// mode 0: Q  -> q_bf  [B,H,N,HC]
// mode 1: KV -> k_bf / v_bf [B,H,PP,HC]
// mode 2: O  -> f32 d_out [M, 768]
// ---------------------------------------------------------------------------
#define GBM 64
#define GBN 128
#define GBK 32
#define ASTRIDE 40   // 80B rows -> 16B aligned vector stores
#define WSTRIDE 34

__global__ __launch_bounds__(256)
void gemm_bf16_kernel(const __bf16* __restrict__ A, const __bf16* __restrict__ W,
                      const float* __restrict__ bias, int M, int K, int Nc,
                      int mode, __bf16* __restrict__ obf0, __bf16* __restrict__ obf1,
                      float* __restrict__ of32) {
  __shared__ __align__(16) __bf16 As[GBM][ASTRIDE];
  __shared__ __bf16 Wt[GBN][WSTRIDE];   // transposed: Wt[n][k]
  int t = threadIdx.x, lane = t & 31, w = t >> 5;
  int wm = w & 3, wn = w >> 2;
  int m0 = blockIdx.y * GBM, n0 = blockIdx.x * GBN;

  // branch-free A row (garbage rows are discarded by the guarded epilogue)
  int ar = t >> 2;                      // 0..63
  int ac = (t & 3) * 8;                 // 0,8,16,24
  int agm = m0 + ar; if (agm > M - 1) agm = M - 1;
  const __bf16* asrc = A + (size_t)agm * K + ac;
  int wk = t >> 4;                      // 0..15
  int wn8 = (t & 15) * 8;               // 0..120

  v8f acc[4];
#pragma unroll
  for (int ct = 0; ct < 4; ++ct) acc[ct] = zero8();

  for (int kk = 0; kk < K; kk += GBK) {
    // A tile 64x32: one b128 load + one b128 LDS store per thread
    BF8 av; av.u = *(const uint4*)(asrc + kk);
    *(uint4*)&As[ar][ac] = av.u;
    // W tile 32x128 (transposed into LDS): two b128 loads + bf16 scatter
#pragma unroll
    for (int i = 0; i < 2; ++i) {
      int kq = wk + i * 16;
      BF8 wv; wv.u = *(const uint4*)(W + (size_t)(kk + kq) * Nc + n0 + wn8);
#pragma unroll
      for (int u = 0; u < 8; ++u) Wt[wn8 + u][kq] = wv.h[u];
    }
    if (kk + GBK < K) {
      __builtin_prefetch(asrc + kk + GBK, 0, 1);
      __builtin_prefetch(W + (size_t)(kk + GBK + wk) * Nc + n0 + wn8, 0, 1);
    }
    __syncthreads();
    v16bf af = frag_a(&As[wm * 16 + (lane & 15)][0]);
    int kb2 = (lane & 16) ? 16 : 0;
#pragma unroll
    for (int ct = 0; ct < 4; ++ct) {
      v16bf bf = frag_b(&Wt[wn * 64 + ct * 16 + (lane & 15)][kb2]);
      acc[ct] = wmma_bf16(af, bf, acc[ct]);
    }
    __syncthreads();
  }

  int mr_base = m0 + wm * 16 + ((lane & 16) ? 8 : 0);
#pragma unroll
  for (int ct = 0; ct < 4; ++ct) {
    int nc = n0 + wn * 64 + ct * 16 + (lane & 15);
    float bv = bias[nc];
#pragma unroll
    for (int j = 0; j < 8; ++j) {
      int mr = mr_base + j;
      if (mr >= M) continue;
      float vv = acc[ct][j] + bv;
      if (mode == 0) {
        int b = mr / N_, nn = mr - b * N_;
        int h = nc >> 6, c = nc & 63;
        obf0[((((size_t)b * H_ + h) * N_ + nn) << 6) + c] = (__bf16)vv;
      } else if (mode == 1) {
        int b = mr / P_, p = mr - b * P_;
        if (nc < DIM_) {
          int h = nc >> 6, c = nc & 63;
          obf0[((((size_t)b * H_ + h) * PP_ + p) << 6) + c] = (__bf16)vv;
        } else {
          int n2 = nc - DIM_, h = n2 >> 6, c = n2 & 63;
          obf1[((((size_t)b * H_ + h) * PP_ + p) << 6) + c] = (__bf16)vv;
        }
      } else {
        of32[(size_t)mr * DIM_ + nc] = vv;
      }
    }
  }
}

// ---------------------------------------------------------------------------
// attention helpers
// ---------------------------------------------------------------------------
#define KSTRIDE 72   // 144B rows -> 16B aligned vector stores

// stage K tile [32 p][64 c] with b128 loads/stores (block = 128 threads)
__device__ inline void stage_k(__bf16 (*ks)[KSTRIDE], const __bf16* __restrict__ kbf,
                               size_t base, int p0, int t) {
  int r = t >> 3;              // 0..15
  int c = (t & 7) * 8;         // 0..56
#pragma unroll
  for (int i = 0; i < 2; ++i) {
    int p = r + i * 16;
    BF8 v; v.u = *(const uint4*)(kbf + base + (size_t)(p0 + p) * HC_ + c);
    *(uint4*)&ks[p][c] = v.u;
  }
}

// score tile helper: S[16 rows n][32 cols p] for one (b,h), via 4 WMMAs
__device__ inline void score_tiles(v16bf aq0, v16bf aq1, const __bf16 (*ks)[KSTRIDE],
                                   const float* __restrict__ pos, int h, int n0,
                                   int p0, float S0[8], float S1[8]) {
  int lane = threadIdx.x & 31;
  int nl = lane & 15;
  int kb2 = (lane & 16) ? 16 : 0;
  v8f s0 = zero8(), s1 = zero8();
#pragma unroll
  for (int kst = 0; kst < 2; ++kst) {
    v16bf aq = kst ? aq1 : aq0;
    v16bf b0 = frag_b(&ks[nl][kb2 + kst * 32]);
    v16bf b1 = frag_b(&ks[16 + nl][kb2 + kst * 32]);
    s0 = wmma_bf16(aq, b0, s0);
    s1 = wmma_bf16(aq, b1, s1);
  }
  int pc0 = p0 + nl, pc1 = pc0 + 16;
#pragma unroll
  for (int j = 0; j < 8; ++j) {
    int nr = n0 + ((lane & 16) ? j + 8 : j);
    if (nr > N_ - 1) nr = N_ - 1;
    size_t pb = ((size_t)h * N_ + nr) * P_;
    S0[j] = (pc0 < P_) ? (s0[j] * SCALE_ + pos[pb + pc0]) : -INFINITY;
    S1[j] = (pc1 < P_) ? (s1[j] * SCALE_ + pos[pb + pc1]) : -INFINITY;
  }
}

// ---------------------------------------------------------------------------
// pass A: flash attention (row softmax + P@V) and column max of raw scores
// grid (B*H, 13), block 128 (4 waves x 16 N-rows)
// ---------------------------------------------------------------------------
__global__ __launch_bounds__(128)
void attn_kernel(const __bf16* __restrict__ qbf, const __bf16* __restrict__ kbf,
                 const __bf16* __restrict__ vbf, const float* __restrict__ pos,
                 float* __restrict__ out_f, float* __restrict__ colmax) {
  __shared__ __align__(16) __bf16 ks[32][KSTRIDE];
  __shared__ __bf16 vts[64][34];       // transposed V tile: vts[c][p]
  __shared__ __bf16 pr[4][16][34];     // wave-private probs (DS in-order per wave)
  int t = threadIdx.x, lane = t & 31, w = t >> 5;
  int bh = blockIdx.x, b = bh / H_, h = bh - b * H_;
  int n0 = blockIdx.y * 64 + w * 16;
  size_t qbase = (size_t)bh * N_ * HC_;
  size_t kvbase = (size_t)bh * PP_ * HC_;

  int qrow = n0 + (lane & 15); if (qrow > N_ - 1) qrow = N_ - 1;
  const __bf16* qr = qbf + qbase + (size_t)qrow * HC_;
  v16bf aq0 = frag_a(qr);
  v16bf aq1 = frag_a(qr + 32);

  v8f acc[4];
#pragma unroll
  for (int ct = 0; ct < 4; ++ct) acc[ct] = zero8();
  float mrow[8], lrow[8];
#pragma unroll
  for (int j = 0; j < 8; ++j) { mrow[j] = -INFINITY; lrow[j] = 0.f; }

  int nl = lane & 15, kb2 = (lane & 16) ? 16 : 0;
  int vr = t >> 3, vc = (t & 7) * 8;

  for (int p0 = 0; p0 < PP_; p0 += 32) {
    __syncthreads();
    stage_k(ks, kbf, kvbase, p0, t);
#pragma unroll
    for (int i = 0; i < 2; ++i) {                // stage transposed V (wide loads)
      int p = vr + i * 16;
      BF8 v; v.u = *(const uint4*)(vbf + kvbase + (size_t)(p0 + p) * HC_ + vc);
#pragma unroll
      for (int u = 0; u < 8; ++u) vts[vc + u][p] = v.h[u];
    }
    if (p0 + 32 < PP_) {                         // prefetch next K/V tiles
      __builtin_prefetch(kbf + kvbase + (size_t)(p0 + 32) * HC_ + t * 16, 0, 1);
      __builtin_prefetch(vbf + kvbase + (size_t)(p0 + 32) * HC_ + t * 16, 0, 1);
    }
    __syncthreads();

    float S0[8], S1[8];
    score_tiles(aq0, aq1, ks, pos, h, n0, p0, S0, S1);

    // column max of raw scores (same column in both 16-lane halves)
    float cm0 = -INFINITY, cm1 = -INFINITY;
#pragma unroll
    for (int j = 0; j < 8; ++j) { cm0 = fmaxf(cm0, S0[j]); cm1 = fmaxf(cm1, S1[j]); }
    cm0 = fmaxf(cm0, __shfl_xor(cm0, 16, 32));
    cm1 = fmaxf(cm1, __shfl_xor(cm1, 16, 32));
    int pc0 = p0 + nl, pc1 = pc0 + 16;
    if (lane < 16) {
      if (pc0 < P_) atomicMaxF(&colmax[(size_t)bh * PP_ + pc0], cm0);
      if (pc1 < P_) atomicMaxF(&colmax[(size_t)bh * PP_ + pc1], cm1);
    }

    // online row softmax + probs to LDS in A-layout source form
#pragma unroll
    for (int j = 0; j < 8; ++j) {
      float tm = fmaxf(S0[j], S1[j]);
      tm = wred_max16(tm);
      float mnew = fmaxf(mrow[j], tm);
      float al = __expf(mrow[j] - mnew);
      float e0 = __expf(S0[j] - mnew);
      float e1 = __expf(S1[j] - mnew);
      float rs = wred_sum16(e0 + e1);
      lrow[j] = lrow[j] * al + rs;
      mrow[j] = mnew;
#pragma unroll
      for (int ct = 0; ct < 4; ++ct) acc[ct][j] *= al;
      int r = (lane & 16) ? j + 8 : j;
      pr[w][r][nl] = (__bf16)e0;
      pr[w][r][16 + nl] = (__bf16)e1;
    }
    v16bf ap = frag_a(&pr[w][nl][0]);            // probs A-fragment [16x32]
#pragma unroll
    for (int ct = 0; ct < 4; ++ct) {
      v16bf bv = frag_b(&vts[ct * 16 + nl][kb2]);
      acc[ct] = wmma_bf16(ap, bv, acc[ct]);
    }
  }

#pragma unroll
  for (int ct = 0; ct < 4; ++ct) {
#pragma unroll
    for (int j = 0; j < 8; ++j) {
      int nr = n0 + ((lane & 16) ? j + 8 : j);
      if (nr < N_)
        out_f[(((size_t)b * N_ + nr) * H_ + h) * HC_ + ct * 16 + nl] =
            acc[ct][j] / lrow[j];
    }
  }
}

// ---------------------------------------------------------------------------
// pass B: column sums of exp(S - colmax)  (softmax over N axis)
// ---------------------------------------------------------------------------
__global__ __launch_bounds__(128)
void colsum_kernel(const __bf16* __restrict__ qbf, const __bf16* __restrict__ kbf,
                   const float* __restrict__ pos, const float* __restrict__ colmax,
                   float* __restrict__ colsum) {
  __shared__ __align__(16) __bf16 ks[32][KSTRIDE];
  int t = threadIdx.x, lane = t & 31, w = t >> 5;
  int bh = blockIdx.x, h = bh % H_;
  int n0 = blockIdx.y * 64 + w * 16;
  size_t qbase = (size_t)bh * N_ * HC_;
  size_t kvbase = (size_t)bh * PP_ * HC_;
  int nl = lane & 15;

  int qrow = n0 + nl; if (qrow > N_ - 1) qrow = N_ - 1;
  const __bf16* qr = qbf + qbase + (size_t)qrow * HC_;
  v16bf aq0 = frag_a(qr);
  v16bf aq1 = frag_a(qr + 32);

  for (int p0 = 0; p0 < PP_; p0 += 32) {
    __syncthreads();
    stage_k(ks, kbf, kvbase, p0, t);
    if (p0 + 32 < PP_)
      __builtin_prefetch(kbf + kvbase + (size_t)(p0 + 32) * HC_ + t * 16, 0, 1);
    __syncthreads();
    float S0[8], S1[8];
    score_tiles(aq0, aq1, ks, pos, h, n0, p0, S0, S1);
    int pc0 = p0 + nl, pc1 = pc0 + 16;
    float cm0 = (pc0 < P_) ? colmax[(size_t)bh * PP_ + pc0] : 0.f;
    float cm1 = (pc1 < P_) ? colmax[(size_t)bh * PP_ + pc1] : 0.f;
    float cs0 = 0.f, cs1 = 0.f;
#pragma unroll
    for (int j = 0; j < 8; ++j) {
      int nr = n0 + ((lane & 16) ? j + 8 : j);
      bool rv = nr < N_;
      cs0 += (rv && pc0 < P_) ? __expf(S0[j] - cm0) : 0.f;
      cs1 += (rv && pc1 < P_) ? __expf(S1[j] - cm1) : 0.f;
    }
    cs0 += __shfl_xor(cs0, 16, 32);
    cs1 += __shfl_xor(cs1, 16, 32);
    if (lane < 16) {
      if (pc0 < P_) atomicAdd(&colsum[(size_t)bh * PP_ + pc0], cs0);
      if (pc1 < P_) atomicAdd(&colsum[(size_t)bh * PP_ + pc1], cs1);
    }
  }
}

// ---------------------------------------------------------------------------
// pass C: fw_acc[b,n] += (1/H) * sum_p exp(S-cmax)/csum * pa[b,p]
// ---------------------------------------------------------------------------
__global__ __launch_bounds__(128)
void fw_kernel(const __bf16* __restrict__ qbf, const __bf16* __restrict__ kbf,
               const float* __restrict__ pos, const float* __restrict__ colmax,
               const float* __restrict__ colsum, const float* __restrict__ pa,
               float* __restrict__ fw_acc) {
  __shared__ __align__(16) __bf16 ks[32][KSTRIDE];
  int t = threadIdx.x, lane = t & 31, w = t >> 5;
  int bh = blockIdx.x, b = bh / H_, h = bh - b * H_;
  int n0 = blockIdx.y * 64 + w * 16;
  size_t qbase = (size_t)bh * N_ * HC_;
  size_t kvbase = (size_t)bh * PP_ * HC_;
  int nl = lane & 15;

  int qrow = n0 + nl; if (qrow > N_ - 1) qrow = N_ - 1;
  const __bf16* qr = qbf + qbase + (size_t)qrow * HC_;
  v16bf aq0 = frag_a(qr);
  v16bf aq1 = frag_a(qr + 32);

  float rs[8];
#pragma unroll
  for (int j = 0; j < 8; ++j) rs[j] = 0.f;

  for (int p0 = 0; p0 < PP_; p0 += 32) {
    __syncthreads();
    stage_k(ks, kbf, kvbase, p0, t);
    if (p0 + 32 < PP_)
      __builtin_prefetch(kbf + kvbase + (size_t)(p0 + 32) * HC_ + t * 16, 0, 1);
    __syncthreads();
    float S0[8], S1[8];
    score_tiles(aq0, aq1, ks, pos, h, n0, p0, S0, S1);
    int pc0 = p0 + nl, pc1 = pc0 + 16;
    float cm0 = (pc0 < P_) ? colmax[(size_t)bh * PP_ + pc0] : 0.f;
    float cm1 = (pc1 < P_) ? colmax[(size_t)bh * PP_ + pc1] : 0.f;
    float f0 = (pc0 < P_) ? pa[b * P_ + pc0] / colsum[(size_t)bh * PP_ + pc0] : 0.f;
    float f1 = (pc1 < P_) ? pa[b * P_ + pc1] / colsum[(size_t)bh * PP_ + pc1] : 0.f;
#pragma unroll
    for (int j = 0; j < 8; ++j) {
      float e0 = (pc0 < P_) ? __expf(S0[j] - cm0) : 0.f;
      float e1 = (pc1 < P_) ? __expf(S1[j] - cm1) : 0.f;
      rs[j] += e0 * f0 + e1 * f1;
    }
  }
#pragma unroll
  for (int j = 0; j < 8; ++j) rs[j] = wred_sum16(rs[j]);
  if ((lane & 15) == 0) {
#pragma unroll
    for (int j = 0; j < 8; ++j) {
      int nr = n0 + ((lane & 16) ? j + 8 : j);
      if (nr < N_) atomicAdd(&fw_acc[b * N_ + nr], rs[j] * (1.f / (float)H_));
    }
  }
}

// ---------------------------------------------------------------------------
// PartSE: gap -> fc -> LN -> gelu -> fc -> sigmoid (tiny)
// ---------------------------------------------------------------------------
__global__ __launch_bounds__(256)
void pmean_kernel(const float* __restrict__ parts, const int* __restrict__ ridx,
                  float* __restrict__ pmean) {
  __shared__ float red[256];
  int id = blockIdx.x, t = threadIdx.x;
  int b = id / P_, p = id - b * P_;
  const float* row = parts + ((size_t)b * PF_ + ridx[p]) * DIM_;
  float s = 0.f;
  for (int c = t; c < DIM_; c += 256) s += row[c];
  red[t] = s; __syncthreads();
  for (int st = 128; st > 0; st >>= 1) { if (t < st) red[t] += red[t + st]; __syncthreads(); }
  if (t == 0) pmean[id] = red[0] * (1.f / (float)DIM_);
}

__global__ __launch_bounds__(512)
void part_se_kernel(const float* __restrict__ pmean, const float* __restrict__ w1,
                    const float* __restrict__ b1, const float* __restrict__ g,
                    const float* __restrict__ beta, const float* __restrict__ w2,
                    const float* __restrict__ b2, float* __restrict__ pa) {
  __shared__ float hin[P_];
  __shared__ float hmid[P_];
  __shared__ float red[512];
  int t = threadIdx.x, b = blockIdx.x;
  if (t < P_) hin[t] = pmean[b * P_ + t];
  __syncthreads();
  if (t < P_) {
    float s = b1[t];
    for (int i = 0; i < P_; ++i) s += hin[i] * w1[i * P_ + t];
    hmid[t] = s;
  }
  __syncthreads();
  red[t] = (t < P_) ? hmid[t] : 0.f; __syncthreads();
  for (int st = 256; st > 0; st >>= 1) { if (t < st) red[t] += red[t + st]; __syncthreads(); }
  float mu = red[0] * (1.f / (float)P_);
  __syncthreads();
  float d = (t < P_) ? (hmid[t] - mu) : 0.f;
  red[t] = d * d; __syncthreads();
  for (int st = 256; st > 0; st >>= 1) { if (t < st) red[t] += red[t + st]; __syncthreads(); }
  float var = red[0] * (1.f / (float)P_);
  __syncthreads();
  if (t < P_) {
    float x = (hmid[t] - mu) * rsqrtf(var + LN_EPS_) * g[t] + beta[t];
    x = 0.5f * x * (1.f + erff(x * 0.70710678118f));   // exact gelu
    hin[t] = x;
  }
  __syncthreads();
  if (t < P_) {
    float s = b2[t];
    for (int i = 0; i < P_; ++i) s += hin[i] * w2[i * P_ + t];
    pa[b * P_ + t] = 1.f / (1.f + __expf(-s));
  }
}

// ---------------------------------------------------------------------------
// finalize: global min/max of fw, then out2 = out + fw_norm*scale (bf16)
// ---------------------------------------------------------------------------
__global__ __launch_bounds__(256)
void minmax_kernel(const float* __restrict__ fw, float* __restrict__ mm) {
  __shared__ float smn[256], smx[256];
  int t = threadIdx.x;
  float mn = INFINITY, mx = -INFINITY;
  for (int i = t; i < BN_; i += 256) {
    float v = fw[i];
    mn = fminf(mn, v); mx = fmaxf(mx, v);
  }
  smn[t] = mn; smx[t] = mx; __syncthreads();
  for (int st = 128; st > 0; st >>= 1) {
    if (t < st) { smn[t] = fminf(smn[t], smn[t + st]); smx[t] = fmaxf(smx[t], smx[t + st]); }
    __syncthreads();
  }
  if (t == 0) { mm[0] = smn[0]; mm[1] = smx[0]; }
}

__global__ void addfw_kernel(const float* __restrict__ out_f,
                             const float* __restrict__ fw_acc,
                             const float* __restrict__ mm,
                             const float* __restrict__ wscale,
                             __bf16* __restrict__ out2,
                             float* __restrict__ dout_fw) {
  int t = blockIdx.x * blockDim.x + threadIdx.x;   // one per 4 elems
  if (t >= BN_ * (DIM_ / 4)) return;
  int m = t / (DIM_ / 4);
  float mn = mm[0], mx = mm[1];
  float fw = (fw_acc[m] - mn) / (mx - mn);
  float ad = fw * wscale[0];
  F4 v; v.f = ((const float4*)out_f)[t];
  BF4 o;
#pragma unroll
  for (int i = 0; i < 4; ++i) o.h[i] = (__bf16)(v.a[i] + ad);
  ((uint2*)out2)[t] = o.u;
  if ((t - m * (DIM_ / 4)) == 0) dout_fw[m] = fw;
}

// ---------------------------------------------------------------------------
// host launch
// ---------------------------------------------------------------------------
extern "C" void kernel_launch(void* const* d_in, const int* in_sizes, int n_in,
                              void* d_out, int out_size, void* d_ws, size_t ws_size,
                              hipStream_t stream) {
  const float* x       = (const float*)d_in[0];
  const float* parts   = (const float*)d_in[1];
  const int*   ridx    = (const int*)  d_in[2];
  const float* q_w     = (const float*)d_in[3];
  const float* q_b     = (const float*)d_in[4];
  const float* kv_w    = (const float*)d_in[5];
  const float* kv_b    = (const float*)d_in[6];
  const float* o_w     = (const float*)d_in[7];
  const float* o_b     = (const float*)d_in[8];
  const float* pos     = (const float*)d_in[9];
  const float* se_w1   = (const float*)d_in[10];
  const float* se_b1   = (const float*)d_in[11];
  const float* se_g    = (const float*)d_in[12];
  const float* se_beta = (const float*)d_in[13];
  const float* se_w2   = (const float*)d_in[14];
  const float* se_b2   = (const float*)d_in[15];
  const float* wscale  = (const float*)d_in[16];

  char* ws = (char*)d_ws;
  size_t off = 0;
  auto take = [&](size_t bytes) { char* p = ws + off; off = (off + bytes + 255) & ~(size_t)255; return p; };

  __bf16* x_bf     = (__bf16*)take((size_t)BN_ * DIM_ * 2);
  __bf16* parts_bf = (__bf16*)take((size_t)BP_ * DIM_ * 2);
  __bf16* qw_bf    = (__bf16*)take((size_t)DIM_ * DIM_ * 2);
  __bf16* kvw_bf   = (__bf16*)take((size_t)DIM_ * 2 * DIM_ * 2);
  __bf16* ow_bf    = (__bf16*)take((size_t)DIM_ * DIM_ * 2);
  __bf16* q_bf     = (__bf16*)take((size_t)B_ * H_ * N_ * HC_ * 2);
  __bf16* k_bf     = (__bf16*)take((size_t)B_ * H_ * PP_ * HC_ * 2);
  __bf16* v_bf     = (__bf16*)take((size_t)B_ * H_ * PP_ * HC_ * 2);
  float*  out_f    = (float*) take((size_t)BN_ * DIM_ * 4);
  __bf16* out2_bf  = (__bf16*)take((size_t)BN_ * DIM_ * 2);
  float*  pmean    = (float*) take((size_t)BP_ * 4);
  float*  pa       = (float*) take((size_t)BP_ * 4);
  float*  colmax   = (float*) take((size_t)B_ * H_ * PP_ * 4);
  float*  colsum   = (float*) take((size_t)B_ * H_ * PP_ * 4);
  float*  fw_acc   = (float*) take((size_t)BN_ * 4);
  float*  mm       = (float*) take(2 * 4);

  float* out_main = (float*)d_out;
  float* out_fw   = out_main + (size_t)BN_ * DIM_;

  // --- prep ---
  int nx4 = BN_ * DIM_ / 4;
  cvt_bf16x4_kernel<<<(nx4 + 255) / 256, 256, 0, stream>>>(x, x_bf, nx4);
  cvt_bf16x4_kernel<<<(DIM_ * DIM_ / 4 + 255) / 256, 256, 0, stream>>>(q_w, qw_bf, DIM_ * DIM_ / 4);
  cvt_bf16x4_kernel<<<(DIM_ * 2 * DIM_ / 4 + 255) / 256, 256, 0, stream>>>(kv_w, kvw_bf, DIM_ * 2 * DIM_ / 4);
  cvt_bf16x4_kernel<<<(DIM_ * DIM_ / 4 + 255) / 256, 256, 0, stream>>>(o_w, ow_bf, DIM_ * DIM_ / 4);
  gather_parts_kernel<<<(BP_ * DIM_ / 4 + 255) / 256, 256, 0, stream>>>(parts, ridx, parts_bf);
  init_kernel<<<(B_ * H_ * PP_ + 255) / 256, 256, 0, stream>>>(colmax, colsum, fw_acc, k_bf, v_bf);

  // --- projections ---
  gemm_bf16_kernel<<<dim3(DIM_ / GBN, (BN_ + GBM - 1) / GBM), 256, 0, stream>>>(
      x_bf, qw_bf, q_b, BN_, DIM_, DIM_, 0, q_bf, nullptr, nullptr);
  gemm_bf16_kernel<<<dim3(2 * DIM_ / GBN, (BP_ + GBM - 1) / GBM), 256, 0, stream>>>(
      parts_bf, kvw_bf, kv_b, BP_, DIM_, 2 * DIM_, 1, k_bf, v_bf, nullptr);

  // --- PartSE ---
  pmean_kernel<<<BP_, 256, 0, stream>>>(parts, ridx, pmean);
  part_se_kernel<<<B_, 512, 0, stream>>>(pmean, se_w1, se_b1, se_g, se_beta, se_w2, se_b2, pa);

  // --- attention ---
  dim3 agrid(B_ * H_, (N_ + 63) / 64);
  attn_kernel<<<agrid, 128, 0, stream>>>(q_bf, k_bf, v_bf, pos, out_f, colmax);
  colsum_kernel<<<agrid, 128, 0, stream>>>(q_bf, k_bf, pos, colmax, colsum);
  fw_kernel<<<agrid, 128, 0, stream>>>(q_bf, k_bf, pos, colmax, colsum, pa, fw_acc);

  // --- finalize ---
  minmax_kernel<<<1, 256, 0, stream>>>(fw_acc, mm);
  addfw_kernel<<<(nx4 + 255) / 256, 256, 0, stream>>>(out_f, fw_acc, mm, wscale, out2_bf, out_fw);
  gemm_bf16_kernel<<<dim3(DIM_ / GBN, (BN_ + GBM - 1) / GBM), 256, 0, stream>>>(
      out2_bf, ow_bf, o_b, BN_, DIM_, DIM_, 2, nullptr, nullptr, out_main);

  (void)in_sizes; (void)n_in; (void)out_size; (void)ws_size;
}
